// MEGNetBlock_1855425871941
// MI455X (gfx1250) — matrix operand order repaired and checked by reading, not compiled, not run on hardware
//
#include <hip/hip_runtime.h>

// ---------------------------------------------------------------------------
// MEGNet block for MI455X (gfx1250): all GEMMs via v_wmma_f32_16x16x32_bf16,
// intermediates kept in LDS (320KB/WGP), edge scatter via global f32 atomics.
// Weight tiles staged into LDS with CDNA5 async-to-LDS loads (ASYNCcnt);
// blocks loop over 4 tiles so weights are staged 4x less often.
// ---------------------------------------------------------------------------

typedef __attribute__((ext_vector_type(16))) __bf16         v16bf;
typedef __attribute__((ext_vector_type(8)))  float          v8f;
typedef __attribute__((ext_vector_type(8)))  unsigned short ushort8;

static constexpr int Bb = 16, Nn = 4096, Ee = 65536;

// ---------------- bf16 helpers -------------------------------------------
__device__ __forceinline__ unsigned short f2bf(float x) {
    union { float f; unsigned u; } v; v.f = x;
    unsigned r = v.u + 0x7fffu + ((v.u >> 16) & 1u);   // round-to-nearest-even
    return (unsigned short)(r >> 16);
}

__device__ __forceinline__ v16bf pack2(ushort8 lo, ushort8 hi) {
    union { struct { ushort8 l, h; } s; v16bf v; } u;
    u.s.l = lo; u.s.h = hi;
    return u.v;
}

// ---------------- WMMA fragment helpers ------------------------------------
// A-frag layout (16x32 bf16): lane r=lane&15 is row, kb=(lane>>4)*8;
//   elems[0..7] = k kb..kb+7, elems[8..15] = k kb+16..kb+23 (two 16B loads).
// B-frag layout (32x16 bf16): lane n=lane&15 is column, kb=(lane>>4)*8;
//   elems[0..7] = k kb..kb+7, elems[8..15] = k kb+16..kb+23 (two 16B loads).
// C layout (16x16 f32): elem v -> row v+8*(lane>>4), col lane&15.
template <int K>
__device__ __forceinline__ void load_afrags(const unsigned short* A, int lane,
                                            v16bf* afr) {
    const int r   = lane & 15;
    const int sel = lane >> 4;
#pragma unroll
    for (int kg = 0; kg < K / 32; ++kg) {
        const int k0 = kg * 32;
        ushort8 alo = *(const ushort8*)(A + r * K + k0 + sel * 8);
        ushort8 ahi = *(const ushort8*)(A + r * K + k0 + sel * 8 + 16);
        afr[kg] = pack2(alo, ahi);
    }
}

template <int K>
__device__ __forceinline__ v8f gemm_tile_c(const v16bf* afr,
                                           const unsigned short* W,
                                           float bv, int lane) {
    const int n   = lane & 15;
    const int sel = lane >> 4;
    v8f c;
#pragma unroll
    for (int i = 0; i < 8; ++i) c[i] = bv;
#pragma unroll
    for (int kg = 0; kg < K / 32; ++kg) {
        const int k0 = kg * 32;
        ushort8 blo = *(const ushort8*)(W + n * K + k0 + sel * 8);
        ushort8 bhi = *(const ushort8*)(W + n * K + k0 + sel * 8 + 16);
        v16bf b = pack2(blo, bhi);
        c = __builtin_amdgcn_wmma_f32_16x16x32_bf16(false, afr[kg], false, b,
                                                    (short)0, c, false, false);
    }
    return c;
}

// Full layer: D(16xOUT, bf16, ld=ldD, starting column col0) = act(A*W^T + bias)
template <int K, int OUT>
__device__ __forceinline__ void gemm_layer(const unsigned short* A,
                                           const unsigned short* W,
                                           const float* bias,
                                           unsigned short* D, int ldD, int col0,
                                           bool relu, int lane) {
    const int n   = lane & 15;
    const int sel = lane >> 4;
    v16bf afr[K / 32];
    load_afrags<K>(A, lane, afr);
#pragma unroll
    for (int og = 0; og < OUT / 16; ++og) {
        v8f c = gemm_tile_c<K>(afr, W + og * 16 * K, bias[og * 16 + n], lane);
#pragma unroll
        for (int v = 0; v < 8; ++v) {
            float x = c[v];
            if (relu) x = fmaxf(x, 0.f);
            D[(v + sel * 8) * ldD + col0 + og * 16 + n] = f2bf(x);
        }
    }
}

// ---- CDNA5 async global->LDS staging (16B granules, ASYNCcnt-tracked) ----
__device__ __forceinline__ void stage_async(unsigned short* dst,
                                            const unsigned short* src,
                                            int n_us, int tid, int nt) {
    const unsigned ldsbase = (unsigned)(size_t)dst;  // addr[31:0] = LDS offset
    const int chunks = n_us / 8;                     // 16B per chunk
    for (int i = tid; i < chunks; i += nt) {
        const void* g = (const void*)(src + i * 8);
        unsigned l = ldsbase + i * 16;
        asm volatile("global_load_async_to_lds_b128 %0, %1, off"
                     :: "v"(l), "v"(g) : "memory");
    }
}

__device__ __forceinline__ void wait_async_all() {
    asm volatile("s_wait_asynccnt 0" ::: "memory");
}

// ---------------- small utility kernels -----------------------------------
__global__ void conv_f32_bf16(const float* __restrict__ s,
                              unsigned short* __restrict__ d, int n) {
    int i = blockIdx.x * blockDim.x + threadIdx.x;
    if (i < n) d[i] = f2bf(s[i]);
}

__global__ void counts_kernel(const int* __restrict__ idx1,
                              float* __restrict__ counts) {
    int e = blockIdx.x * blockDim.x + threadIdx.x;
    if (e < Ee) atomicAdd(&counts[idx1[e]], 1.0f);
}

// ---------------- state pre-MLP: st1 = pre(states)   (one wave, M = B = 16) --
__global__ __launch_bounds__(32)
void st1_kernel(const float* __restrict__ states,
                const unsigned short* __restrict__ gW0, const float* __restrict__ b0,
                const unsigned short* __restrict__ gW1, const float* __restrict__ b1,
                float* __restrict__ st1out) {
    __shared__ unsigned short sIn[16 * 64];
    __shared__ unsigned short sH[16 * 128];
    const int lane = threadIdx.x;
    for (int i = lane; i < 16 * 64; i += 32) sIn[i] = f2bf(states[i]);
    gemm_layer<64, 128>(sIn, gW0, b0, sH, 128, 0, true, lane);
    const int n = lane & 15, sel = lane >> 4;
    v16bf afr[4];
    load_afrags<128>(sH, lane, afr);
#pragma unroll
    for (int og = 0; og < 4; ++og) {
        v8f c = gemm_tile_c<128>(afr, gW1 + og * 16 * 128, b1[og * 16 + n], lane);
#pragma unroll
        for (int v = 0; v < 8; ++v)
            st1out[(v + sel * 8) * 64 + og * 16 + n] = fmaxf(c[v], 0.f);
    }
}

// ---------------- site pre-MLP: s1 = pre(sites) -> bf16 workspace ----------
static constexpr int S1_SMEM = (8192 + 8192) * 2 + 8 * (1024 + 2048) * 2; // 81920
__global__ __launch_bounds__(256)
void s1_kernel(const float* __restrict__ sites,
               const unsigned short* __restrict__ gW0, const float* __restrict__ b0,
               const unsigned short* __restrict__ gW1, const float* __restrict__ b1,
               unsigned short* __restrict__ s1bf) {
    extern __shared__ __align__(16) unsigned char smem[];
    unsigned short* sW0 = (unsigned short*)smem;            // 128x64
    unsigned short* sW1 = sW0 + 8192;                       // 64x128
    unsigned short* wbase = sW1 + 8192;
    const int tid = threadIdx.x, lane = tid & 31, wv = tid >> 5;
    stage_async(sW0, gW0, 8192, tid, blockDim.x);
    stage_async(sW1, gW1, 8192, tid, blockDim.x);
    wait_async_all();
    __syncthreads();

    unsigned short* sIn = wbase + wv * 3072;                // 16x64
    unsigned short* sH  = sIn + 1024;                       // 16x128
    const size_t r0 = (size_t)(blockIdx.x * 8 + wv) * 16;   // flattened B*N rows

    for (int i = lane; i < 16 * 64; i += 32)
        sIn[i] = f2bf(sites[r0 * 64 + i]);

    gemm_layer<64, 128>(sIn, sW0, b0, sH, 128, 0, true, lane);
    const int n = lane & 15, sel = lane >> 4;
    v16bf afr[4];
    load_afrags<128>(sH, lane, afr);
#pragma unroll
    for (int og = 0; og < 4; ++og) {
        v8f c = gemm_tile_c<128>(afr, sW1 + og * 16 * 128, b1[og * 16 + n], lane);
#pragma unroll
        for (int v = 0; v < 8; ++v)
            s1bf[(r0 + v + sel * 8) * 64 + og * 16 + n] = f2bf(fmaxf(c[v], 0.f));
    }
}

// ---------------- fused edge kernel ----------------------------------------
// per wave-iteration: 16 edges.  x = [g1 | g2 | relu(pre(bonds)) | st1] (16x256)
// b2 = bu_W2( relu(bu_W1( relu(bu_W0 x) )) );  bonds_out = bonds + b2;
// scatter-add b2 into poolsum[b, idx1, :] and batch-sum bsum[b, :].
static constexpr int BOND_W_US = 8192 + 8192 + 32768 + 8192 + 4096 + 64; // 61504
static constexpr int BOND_SMEM =
    BOND_W_US * 2 + 64 * 4 + 128 * 4 * 2 + 8 * 7168 * 2;                 // 238976
__global__ __launch_bounds__(256)
void bond_kernel(const float* __restrict__ bonds,
                 const unsigned short* __restrict__ s1bf,
                 const float* __restrict__ st1,
                 const int* __restrict__ idx1, const int* __restrict__ idx2,
                 const unsigned short* gWb0, const float* __restrict__ bb0,
                 const unsigned short* gWb1, const float* __restrict__ bb1,
                 const unsigned short* gWu0, const float* __restrict__ bub0,
                 const unsigned short* gWu1, const float* __restrict__ bub1,
                 const unsigned short* gWu2, const float* __restrict__ bub2,
                 float* __restrict__ bonds_out,
                 float* __restrict__ poolsum,
                 float* __restrict__ bsum) {
    extern __shared__ __align__(16) unsigned char smem[];
    unsigned short* sWb0 = (unsigned short*)smem;   // 128x64
    unsigned short* sWb1 = sWb0 + 8192;             // 64x128
    unsigned short* sWu0 = sWb1 + 8192;             // 128x256
    unsigned short* sWu1 = sWu0 + 32768;            // 64x128
    unsigned short* sWu2 = sWu1 + 8192;             // 64x64
    unsigned short* sSt  = sWu2 + 4096;             // 64
    float* bsumL = (float*)(sSt + 64);              // 64
    int*   sIdx1 = (int*)(bsumL + 64);              // 128
    int*   sIdx2 = sIdx1 + 128;                     // 128
    unsigned short* wbase = (unsigned short*)(sIdx2 + 128);

    const int tid = threadIdx.x, nt = blockDim.x;
    const int wv = tid >> 5, lane = tid & 31;
    const int b  = blockIdx.y;

    stage_async(sWb0, gWb0, 8192, tid, nt);
    stage_async(sWb1, gWb1, 8192, tid, nt);
    stage_async(sWu0, gWu0, 32768, tid, nt);
    stage_async(sWu1, gWu1, 8192, tid, nt);
    stage_async(sWu2, gWu2, 4096, tid, nt);
    if (tid < 64) { sSt[tid] = f2bf(st1[b * 64 + tid]); bsumL[tid] = 0.f; }
    wait_async_all();
    __syncthreads();

    unsigned short* xbuf = wbase + wv * 7168;   // 16x256 (concat input)
    unsigned short* sH   = xbuf + 4096;         // 16x128
    unsigned short* sIn  = sH + 2048;           // 16x64
    const int n = lane & 15, sel = lane >> 4;

    for (int it = 0; it < 4; ++it) {
        const int e0 = (blockIdx.x * 4 + it) * 128 + wv * 16;

        if (lane < 16) {
            sIdx1[wv * 16 + lane] = idx1[e0 + lane];
            sIdx2[wv * 16 + lane] = idx2[e0 + lane];
        }
        // stage raw bond rows (f32 -> bf16)
        for (int i = lane; i < 16 * 64; i += 32) {
            int row = i >> 6, col = i & 63;
            sIn[i] = f2bf(bonds[((size_t)b * Ee + e0 + row) * 64 + col]);
        }
        // stage gathers g1,g2 (bf16 rows from L2-resident s1) and st broadcast
        {
            unsigned int* xd = (unsigned int*)xbuf;             // 128 dw / row
            const unsigned int* std_ = (const unsigned int*)sSt;
            for (int j = lane; j < 16 * 32; j += 32) {
                int row = j >> 5, cd = j & 31;
                int n1 = sIdx1[wv * 16 + row];
                int n2 = sIdx2[wv * 16 + row];
                const unsigned int* r1 = (const unsigned int*)(s1bf + ((size_t)b * Nn + n1) * 64);
                const unsigned int* r2 = (const unsigned int*)(s1bf + ((size_t)b * Nn + n2) * 64);
                xd[row * 128 + cd]      = r1[cd];               // cols   0..63
                xd[row * 128 + 32 + cd] = r2[cd];               // cols  64..127
                xd[row * 128 + 96 + cd] = std_[cd];             // cols 192..255
            }
        }
        // bond pre-MLP (writes cols 128..191 of xbuf)
        gemm_layer<64, 128>(sIn, sWb0, bb0, sH, 128, 0, true, lane);
        gemm_layer<128, 64>(sH, sWb1, bb1, xbuf, 256, 128, true, lane);
        // bond-update MLP
        gemm_layer<256, 128>(xbuf, sWu0, bub0, sH, 128, 0, true, lane);
        gemm_layer<128, 64>(sH, sWu1, bub1, sIn, 64, 0, true, lane);

        v16bf afr[2];
        load_afrags<64>(sIn, lane, afr);
#pragma unroll
        for (int og = 0; og < 4; ++og) {
            v8f c = gemm_tile_c<64>(afr, sWu2 + og * 16 * 64, bub2[og * 16 + n], lane);
            float part = 0.f;
#pragma unroll
            for (int v = 0; v < 8; ++v) {
                const int m   = v + sel * 8;
                const int col = og * 16 + n;
                const float val = c[v];
                const size_t gofs = ((size_t)b * Ee + e0 + m) * 64 + col;
                bonds_out[gofs] = bonds[gofs] + val;
                const int node = sIdx1[wv * 16 + m];
                atomicAdd(&poolsum[((size_t)b * Nn + node) * 64 + col], val);
                part += val;
            }
            atomicAdd(&bsumL[og * 16 + n], part);
        }
    }
    __syncthreads();
    if (tid < 64) atomicAdd(&bsum[b * 64 + tid], bsumL[tid]);
}

// ---------------- fused site kernel ----------------------------------------
static constexpr int SITE_W_US = 24576 + 8192 + 4096 + 64;               // 36928
static constexpr int SITE_SMEM = SITE_W_US * 2 + 64 * 4 + 8 * 6144 * 2;  // 172416
__global__ __launch_bounds__(256)
void site_kernel(const float* __restrict__ sites,
                 const unsigned short* __restrict__ s1bf,
                 const float* __restrict__ st1,
                 const float* __restrict__ poolsum,
                 const float* __restrict__ counts,
                 const unsigned short* gW0, const float* __restrict__ b0,
                 const unsigned short* gW1, const float* __restrict__ b1,
                 const unsigned short* gW2, const float* __restrict__ b2,
                 float* __restrict__ sites_out, float* __restrict__ ssum) {
    extern __shared__ __align__(16) unsigned char smem[];
    unsigned short* sW0 = (unsigned short*)smem;    // 128x192
    unsigned short* sW1 = sW0 + 24576;              // 64x128
    unsigned short* sW2 = sW1 + 8192;               // 64x64
    unsigned short* sSt = sW2 + 4096;               // 64
    float* ssumL = (float*)(sSt + 64);              // 64
    unsigned short* wbase = (unsigned short*)(ssumL + 64);

    const int tid = threadIdx.x, nt = blockDim.x;
    const int wv = tid >> 5, lane = tid & 31;
    const int b  = blockIdx.y;

    stage_async(sW0, gW0, 24576, tid, nt);
    stage_async(sW1, gW1, 8192, tid, nt);
    stage_async(sW2, gW2, 4096, tid, nt);
    if (tid < 64) { sSt[tid] = f2bf(st1[b * 64 + tid]); ssumL[tid] = 0.f; }
    wait_async_all();
    __syncthreads();

    unsigned short* xbuf = wbase + wv * 6144;   // 16x192
    unsigned short* sH   = xbuf + 3072;         // 16x128
    unsigned short* sIn  = sH + 2048;           // 16x64
    const int n = lane & 15, sel = lane >> 4;

    for (int it = 0; it < 4; ++it) {
        const int n0 = (blockIdx.x * 4 + it) * 128 + wv * 16;

        // cols 0..63: pool = poolsum / max(counts,1)
        for (int i = lane; i < 16 * 64; i += 32) {
            int row = i >> 6, col = i & 63;
            int nd = n0 + row;
            float val = poolsum[((size_t)b * Nn + nd) * 64 + col] / fmaxf(counts[nd], 1.f);
            xbuf[row * 192 + col] = f2bf(val);
        }
        // cols 64..127: s1 rows;  cols 128..191: st broadcast
        {
            unsigned int* xd = (unsigned int*)xbuf;   // 96 dw / row
            const unsigned int* std_ = (const unsigned int*)sSt;
            for (int j = lane; j < 16 * 32; j += 32) {
                int row = j >> 5, cd = j & 31;
                const unsigned int* r1 =
                    (const unsigned int*)(s1bf + ((size_t)b * Nn + n0 + row) * 64);
                xd[row * 96 + 32 + cd] = r1[cd];
                xd[row * 96 + 64 + cd] = std_[cd];
            }
        }
        gemm_layer<192, 128>(xbuf, sW0, b0, sH, 128, 0, true, lane);
        gemm_layer<128, 64>(sH, sW1, b1, sIn, 64, 0, true, lane);

        v16bf afr[2];
        load_afrags<64>(sIn, lane, afr);
#pragma unroll
        for (int og = 0; og < 4; ++og) {
            v8f c = gemm_tile_c<64>(afr, sW2 + og * 16 * 64, b2[og * 16 + n], lane);
            float part = 0.f;
#pragma unroll
            for (int v = 0; v < 8; ++v) {
                const int m = v + sel * 8, col = og * 16 + n;
                const float val = fmaxf(c[v], 0.f);          // s2 has final relu
                const size_t gofs = ((size_t)b * Nn + n0 + m) * 64 + col;
                sites_out[gofs] = sites[gofs] + val;
                part += val;
            }
            atomicAdd(&ssumL[og * 16 + n], part);
        }
    }
    __syncthreads();
    if (tid < 64) atomicAdd(&ssum[b * 64 + tid], ssumL[tid]);
}

// ---------------- state-update kernel (one wave, M = B = 16) ----------------
__global__ __launch_bounds__(32)
void state_kernel(const float* __restrict__ states,
                  const float* __restrict__ bsum, const float* __restrict__ ssum,
                  const float* __restrict__ st1,
                  const unsigned short* gW0, const float* __restrict__ b0,
                  const unsigned short* gW1, const float* __restrict__ b1,
                  const unsigned short* gW2, const float* __restrict__ b2,
                  float* __restrict__ states_out) {
    __shared__ unsigned short xbuf[16 * 192];
    __shared__ unsigned short sH[16 * 128];
    __shared__ unsigned short sIn[16 * 64];
    const int lane = threadIdx.x;
    for (int i = lane; i < 16 * 192; i += 32) {
        int row = i / 192, col = i % 192;
        float v;
        if (col < 64)       v = bsum[row * 64 + col] * (1.f / Ee);
        else if (col < 128) v = ssum[row * 64 + col - 64] * (1.f / Nn);
        else                v = st1[row * 64 + col - 128];
        xbuf[i] = f2bf(v);
    }
    gemm_layer<192, 128>(xbuf, gW0, b0, sH, 128, 0, true, lane);
    gemm_layer<128, 64>(sH, gW1, b1, sIn, 64, 0, true, lane);
    const int n = lane & 15, sel = lane >> 4;
    v16bf afr[2];
    load_afrags<64>(sIn, lane, afr);
#pragma unroll
    for (int og = 0; og < 4; ++og) {
        v8f c = gemm_tile_c<64>(afr, gW2 + og * 16 * 64, b2[og * 16 + n], lane);
#pragma unroll
        for (int v = 0; v < 8; ++v) {
            const int m = v + sel * 8, col = og * 16 + n;
            states_out[m * 64 + col] = states[m * 64 + col] + fmaxf(c[v], 0.f);
        }
    }
}

// ---------------------------------------------------------------------------
extern "C" void kernel_launch(void* const* d_in, const int* in_sizes, int n_in,
                              void* d_out, int out_size, void* d_ws, size_t ws_size,
                              hipStream_t stream) {
    (void)in_sizes; (void)n_in; (void)out_size; (void)ws_size;

    const float* sites  = (const float*)d_in[0];
    const float* bonds  = (const float*)d_in[1];
    const float* states = (const float*)d_in[2];
    const int*   idx1   = (const int*)d_in[3];
    const int*   idx2   = (const int*)d_in[4];
    const float* bfc_W0 = (const float*)d_in[5];
    const float* bfc_b0 = (const float*)d_in[6];
    const float* bfc_W1 = (const float*)d_in[7];
    const float* bfc_b1 = (const float*)d_in[8];
    const float* sfc_W0 = (const float*)d_in[9];
    const float* sfc_b0 = (const float*)d_in[10];
    const float* sfc_W1 = (const float*)d_in[11];
    const float* sfc_b1 = (const float*)d_in[12];
    const float* stfc_W0 = (const float*)d_in[13];
    const float* stfc_b0 = (const float*)d_in[14];
    const float* stfc_W1 = (const float*)d_in[15];
    const float* stfc_b1 = (const float*)d_in[16];
    const float* bu_W0 = (const float*)d_in[17];
    const float* bu_b0 = (const float*)d_in[18];
    const float* bu_W1 = (const float*)d_in[19];
    const float* bu_b1 = (const float*)d_in[20];
    const float* bu_W2 = (const float*)d_in[21];
    const float* bu_b2 = (const float*)d_in[22];
    const float* su_W0 = (const float*)d_in[23];
    const float* su_b0 = (const float*)d_in[24];
    const float* su_W1 = (const float*)d_in[25];
    const float* su_b1 = (const float*)d_in[26];
    const float* su_W2 = (const float*)d_in[27];
    const float* su_b2 = (const float*)d_in[28];
    const float* stu_W0 = (const float*)d_in[29];
    const float* stu_b0 = (const float*)d_in[30];
    const float* stu_W1 = (const float*)d_in[31];
    const float* stu_b1 = (const float*)d_in[32];
    const float* stu_W2 = (const float*)d_in[33];
    const float* stu_b2 = (const float*)d_in[34];

    float* sites_out  = (float*)d_out;
    float* bonds_out  = sites_out + (size_t)Bb * Nn * 64;
    float* states_out = bonds_out + (size_t)Bb * Ee * 64;

    // ---- workspace layout (256B aligned regions) ----
    char* ws = (char*)d_ws;
    unsigned short* s1bf = (unsigned short*)ws;               // 8,388,608 B
    float* poolsum = (float*)(ws + 8388608);                  // 16,777,216 B
    float* counts  = (float*)(ws + 25165824);                 // 16,384 B
    float* st1     = (float*)(ws + 25182208);                 // 4,096 B
    float* bsum    = (float*)(ws + 25186304);                 // 4,096 B
    float* ssum    = (float*)(ws + 25190400);                 // 4,096 B
    unsigned short* wreg = (unsigned short*)(ws + 25194496);  // bf16 weights

    unsigned short* wSFC0 = wreg;            // 128x64
    unsigned short* wSFC1 = wreg + 8192;     // 64x128
    unsigned short* wBFC0 = wreg + 16384;
    unsigned short* wBFC1 = wreg + 24576;
    unsigned short* wSTFC0 = wreg + 32768;
    unsigned short* wSTFC1 = wreg + 40960;
    unsigned short* wBU0 = wreg + 49152;     // 128x256
    unsigned short* wBU1 = wreg + 81920;     // 64x128
    unsigned short* wBU2 = wreg + 90112;     // 64x64
    unsigned short* wSU0 = wreg + 94208;     // 128x192
    unsigned short* wSU1 = wreg + 118784;
    unsigned short* wSU2 = wreg + 126976;
    unsigned short* wSTU0 = wreg + 131072;   // 128x192
    unsigned short* wSTU1 = wreg + 155648;
    unsigned short* wSTU2 = wreg + 163840;

    hipFuncSetAttribute(reinterpret_cast<const void*>(bond_kernel),
                        hipFuncAttributeMaxDynamicSharedMemorySize, BOND_SMEM);
    hipFuncSetAttribute(reinterpret_cast<const void*>(site_kernel),
                        hipFuncAttributeMaxDynamicSharedMemorySize, SITE_SMEM);
    hipFuncSetAttribute(reinterpret_cast<const void*>(s1_kernel),
                        hipFuncAttributeMaxDynamicSharedMemorySize, S1_SMEM);

    auto cvt = [&](const float* s, unsigned short* d, int n) {
        conv_f32_bf16<<<(n + 255) / 256, 256, 0, stream>>>(s, d, n);
    };
    cvt(sfc_W0, wSFC0, 8192);   cvt(sfc_W1, wSFC1, 8192);
    cvt(bfc_W0, wBFC0, 8192);   cvt(bfc_W1, wBFC1, 8192);
    cvt(stfc_W0, wSTFC0, 8192); cvt(stfc_W1, wSTFC1, 8192);
    cvt(bu_W0, wBU0, 32768);    cvt(bu_W1, wBU1, 8192);  cvt(bu_W2, wBU2, 4096);
    cvt(su_W0, wSU0, 24576);    cvt(su_W1, wSU1, 8192);  cvt(su_W2, wSU2, 4096);
    cvt(stu_W0, wSTU0, 24576);  cvt(stu_W1, wSTU1, 8192); cvt(stu_W2, wSTU2, 4096);

    hipMemsetAsync(poolsum, 0, (size_t)Bb * Nn * 64 * 4, stream);
    hipMemsetAsync(counts, 0, Nn * 4, stream);
    hipMemsetAsync(bsum, 0, Bb * 64 * 4, stream);
    hipMemsetAsync(ssum, 0, Bb * 64 * 4, stream);

    counts_kernel<<<Ee / 256, 256, 0, stream>>>(idx1, counts);

    st1_kernel<<<1, 32, 0, stream>>>(states, wSTFC0, stfc_b0, wSTFC1, stfc_b1, st1);

    s1_kernel<<<(Bb * Nn) / 128, 256, S1_SMEM, stream>>>(
        sites, wSFC0, sfc_b0, wSFC1, sfc_b1, s1bf);

    bond_kernel<<<dim3(Ee / 512, Bb), 256, BOND_SMEM, stream>>>(
        bonds, s1bf, st1, idx1, idx2,
        wBFC0, bfc_b0, wBFC1, bfc_b1,
        wBU0, bu_b0, wBU1, bu_b1, wBU2, bu_b2,
        bonds_out, poolsum, bsum);

    site_kernel<<<dim3(Nn / 512, Bb), 256, SITE_SMEM, stream>>>(
        sites, s1bf, st1, poolsum, counts,
        wSU0, su_b0, wSU1, su_b1, wSU2, su_b2,
        sites_out, ssum);

    state_kernel<<<1, 32, 0, stream>>>(
        states, bsum, ssum, st1,
        wSTU0, stu_b0, wSTU1, stu_b1, wSTU2, stu_b2,
        states_out);
}
